// PolynomialAttn_56530359550028
// MI455X (gfx1250) — matrix operand order repaired and checked by reading, not compiled, not run on hardware
//
#include <hip/hip_runtime.h>

typedef __bf16 v16bf __attribute__((ext_vector_type(16)));
typedef __bf16 v8bf  __attribute__((ext_vector_type(8)));
typedef float  v8f   __attribute__((ext_vector_type(8)));

#define B_ 2
#define H_ 16
#define S_ 2048
#define D_ 64
#define EPS_ 1e-4f

constexpr int WAVES  = 8;   // 256 threads = 8 wave32
constexpr int QTILE  = 16;  // rows per WMMA tile
constexpr int UTILES = 2;   // Q tiles per wave -> 32 query rows per wave
constexpr int TT     = 64;  // key/value rows staged in LDS per iteration
constexpr int KROW   = 80;  // bf16 row stride for K/V tiles (160B -> 32B aligned rows)
constexpr int PROW   = 40;  // bf16 row stride for P staging (80B -> 16B aligned halves)

static __device__ __forceinline__ unsigned fbits(float f) {
  return __builtin_bit_cast(unsigned, f);
}
// fp32 -> bf16 by truncation: "store high 16 bits" folds into ds_store_b16_d16_hi
static __device__ __forceinline__ __bf16 f2bf(float f) {
  return __builtin_bit_cast(__bf16, (unsigned short)(fbits(f) >> 16));
}
// pack two fp32 high-halves into {lo=bf16(a), hi=bf16(b)} with one v_perm_b32
static __device__ __forceinline__ unsigned pack2bf(float a, float b) {
  return __builtin_amdgcn_perm(fbits(b), fbits(a), 0x07060302u);
}

struct BfPair { v8bf lo, hi; };  // 32B, bit-castable to v16bf

// build a v8bf (8 bf16) from 8 contiguous fp32 via 2 b128 loads + 4 v_perm
static __device__ __forceinline__ v8bf load8_bf(const float* p) {
  const float4 a = *(const float4*)p;
  const float4 b = *(const float4*)(p + 4);
  uint4 u;
  u.x = pack2bf(a.x, a.y);
  u.y = pack2bf(a.z, a.w);
  u.z = pack2bf(b.x, b.y);
  u.w = pack2bf(b.z, b.w);
  return __builtin_bit_cast(v8bf, u);
}

__global__ __launch_bounds__(256, 2)
void poly_attn_wmma(const float* __restrict__ Q, const float* __restrict__ K,
                    const float* __restrict__ V, float* __restrict__ O) {
  __shared__ alignas(32) __bf16 Klds[TT][KROW];           // K tile, row-major [t][d]
  __shared__ alignas(32) __bf16 Vt[D_][KROW];             // V tile, transposed [d][t]
  __shared__ alignas(32) __bf16 Plds[WAVES][QTILE][PROW]; // wave-private P re-layout

  const int tid  = threadIdx.x;
  const int wave = tid >> 5;
  const int lane = tid & 31;
  const int l15  = lane & 15;
  const int hi   = lane >> 4;

  const int bh    = blockIdx.x;
  const int qbase = blockIdx.y * (WAVES * UTILES * QTILE) + wave * (UTILES * QTILE);

  const float* Qg = Q + (size_t)bh * S_ * D_;
  const float* Kg = K + (size_t)bh * S_ * D_;
  const float* Vg = V + (size_t)bh * S_ * D_;
  float*       Og = O + (size_t)bh * S_ * D_;

  // ---- Q A-fragments: qa[u][f], tile u rows [qbase+16u, +16), f covers d in [32f, 32f+32)
  // A k order used here (16-bit 16x32): lane m = L&15; elem i -> K = hi*16 + i (contiguous, matches B)
  v16bf qa[UTILES][2];
#pragma unroll
  for (int u = 0; u < UTILES; ++u) {
#pragma unroll
    for (int f = 0; f < 2; ++f) {
      const float* qrow = Qg + (size_t)(qbase + u * QTILE + l15) * D_ + f * 32;
      BfPair qp;
      qp.lo = load8_bf(qrow + hi * 16);
      qp.hi = load8_bf(qrow + hi * 16 + 8);
      qa[u][f] = __builtin_bit_cast(v16bf, qp);
    }
  }

  // all-ones B fragment (layout-independent) for the denominator GEMM
  const __bf16 one = __builtin_bit_cast(__bf16, (unsigned short)0x3F80);
  v16bf ones;
#pragma unroll
  for (int i = 0; i < 16; ++i) ones[i] = one;

  v8f oacc[UTILES][4];  // O accumulators: 2 x (16 rows x 64 cols)
  v8f dacc[UTILES];     // denominator accumulators
#pragma unroll
  for (int u = 0; u < UTILES; ++u) {
#pragma unroll
    for (int r = 0; r < 8; ++r) {
      dacc[u][r] = 0.f;
#pragma unroll
      for (int dc = 0; dc < 4; ++dc) oacc[u][dc][r] = 0.f;
    }
  }

  for (int t0 = 0; t0 < S_; t0 += TT) {
    __syncthreads();
    // cooperative stage: issue all global_load_b128 first, then convert/store.
    float4 kf[4], vf[4];
#pragma unroll
    for (int it = 0; it < 4; ++it) {
      const int idx = tid + it * 256;      // 1024 float4s = 64x64 tile
      const int r   = idx >> 4;
      const int c4  = (idx & 15) * 4;
      kf[it] = *(const float4*)&Kg[(size_t)(t0 + r) * D_ + c4];
      vf[it] = *(const float4*)&Vg[(size_t)(t0 + r) * D_ + c4];
    }
#pragma unroll
    for (int it = 0; it < 4; ++it) {
      const int idx = tid + it * 256;
      const int r   = idx >> 4;
      const int c4  = (idx & 15) * 4;
      uint2 kp;
      kp.x = pack2bf(kf[it].x, kf[it].y);
      kp.y = pack2bf(kf[it].z, kf[it].w);
      *(uint2*)&Klds[r][c4] = kp;          // 4 bf16 in one 8B store
      Vt[c4 + 0][r] = f2bf(vf[it].x);      // transposed via free d16_hi stores
      Vt[c4 + 1][r] = f2bf(vf[it].y);
      Vt[c4 + 2][r] = f2bf(vf[it].z);
      Vt[c4 + 3][r] = f2bf(vf[it].w);
    }
    __syncthreads();

#pragma unroll
    for (int tc = 0; tc < TT; tc += 32) {
      // ---- load K/V B-fragments ONCE, reuse across both Q tiles
      // B k order used here (16-bit 32x16): lane n = L&15; elem i -> K = hi*16 + i
      v16bf kb[2][2];  // [st][f]: B[k][n] = K[tc+16st+n][32f + k]
#pragma unroll
      for (int st = 0; st < 2; ++st)
#pragma unroll
        for (int f = 0; f < 2; ++f)
          kb[st][f] = *(const v16bf*)&Klds[tc + st * 16 + l15][f * 32 + hi * 16];
      v16bf vb[4];     // [dc]: B[k][n] = V[t0+tc+k][16dc+n] = Vt[16dc+n][tc+k]
#pragma unroll
      for (int dc = 0; dc < 4; ++dc)
        vb[dc] = *(const v16bf*)&Vt[dc * 16 + l15][tc + hi * 16];

#pragma unroll
      for (int u = 0; u < UTILES; ++u) {
        // ---- S = Q K^T for t columns [tc, tc+32): two 16x16 tiles
#pragma unroll
        for (int st = 0; st < 2; ++st) {
          v8f s;
#pragma unroll
          for (int r = 0; r < 8; ++r) s[r] = 0.f;
          s = __builtin_amdgcn_wmma_f32_16x16x32_bf16(false, qa[u][0], false, kb[st][0],
                                                      (short)0, s, false, false);
          s = __builtin_amdgcn_wmma_f32_16x16x32_bf16(false, qa[u][1], false, kb[st][1],
                                                      (short)0, s, false, false);
          // square (degree-2 kernel); trunc-bf16 store is d16_hi (free)
          // C layout: vgpr r, lane L -> M = r + 8*hi, N = l15
#pragma unroll
          for (int r = 0; r < 8; ++r) {
            const float p = s[r] * s[r];
            Plds[wave][r + 8 * hi][st * 16 + l15] = f2bf(p);
          }
        }
        // ---- reload P (16x32) as an A fragment (same-wave LDS ops stay ordered)
        BfPair pp;
        pp.lo = *(const v8bf*)&Plds[wave][l15][hi * 16];
        pp.hi = *(const v8bf*)&Plds[wave][l15][hi * 16 + 8];
        const v16bf pa = __builtin_bit_cast(v16bf, pp);

        // ---- denominator: rowsum(P) via P @ ones
        dacc[u] = __builtin_amdgcn_wmma_f32_16x16x32_bf16(false, pa, false, ones,
                                                          (short)0, dacc[u], false, false);
        // ---- O += P @ V : 4 N-chunks of 16 output columns
#pragma unroll
        for (int dc = 0; dc < 4; ++dc)
          oacc[u][dc] = __builtin_amdgcn_wmma_f32_16x16x32_bf16(false, pa, false, vb[dc],
                                                                (short)0, oacc[u][dc],
                                                                false, false);
      }
    }
  }

  // ---- normalize and store: O[M][d] * 1/max(denom[M], eps)
#pragma unroll
  for (int u = 0; u < UTILES; ++u) {
#pragma unroll
    for (int r = 0; r < 8; ++r) {
      const float inv  = 1.f / fmaxf(dacc[u][r], EPS_);
      const size_t row = (size_t)(qbase + u * QTILE + r + 8 * hi) * D_;
#pragma unroll
      for (int dc = 0; dc < 4; ++dc)
        Og[row + dc * 16 + l15] = oacc[u][dc][r] * inv;
    }
  }
}

extern "C" void kernel_launch(void* const* d_in, const int* in_sizes, int n_in,
                              void* d_out, int out_size, void* d_ws, size_t ws_size,
                              hipStream_t stream) {
  const float* q = (const float*)d_in[0];
  const float* k = (const float*)d_in[1];
  const float* v = (const float*)d_in[2];
  float* out = (float*)d_out;

  dim3 grid(B_ * H_, S_ / (WAVES * UTILES * QTILE));  // 32 x 8 workgroups
  dim3 block(256);                                    // 8 wave32 per workgroup
  hipLaunchKernelGGL(poly_attn_wmma, grid, block, 0, stream, q, k, v, out);
}
